// MarketGraphNet_43903155699920
// MI455X (gfx1250) — matrix operand order repaired and checked by reading, not compiled, not run on hardware
//
#include <hip/hip_runtime.h>
#include <math.h>

typedef __bf16 bf16_t;
typedef __attribute__((ext_vector_type(16))) __bf16 v16bf;
typedef __attribute__((ext_vector_type(8)))  __bf16 v8bf;
typedef __attribute__((ext_vector_type(8)))  float  v8f;

#define N_NODES 50000
#define N_EDGES 800000

// ---------------- helpers: orderable-uint encoding for float atomic max ----------------
__device__ __forceinline__ unsigned enc_f32(float f) {
  unsigned u = __float_as_uint(f);
  return (u & 0x80000000u) ? ~u : (u | 0x80000000u);
}
__device__ __forceinline__ float dec_f32(unsigned u) {
  return (u & 0x80000000u) ? __uint_as_float(u & 0x7fffffffu) : __uint_as_float(~u);
}
#define ENC_NEG_INF 0x007FFFFFu   // enc(-inf)

// ---------------- generic vectorized fill (n must be a multiple of 4) -------------------
__global__ void fill_u32_kernel(uint4* __restrict__ p, unsigned v, long n4) {
  long i = (long)blockIdx.x * blockDim.x + threadIdx.x;
  if (i < n4) p[i] = make_uint4(v, v, v, v);
}

// ---------------- WMMA fused linear: OUT = [relu]( A0[/den] @ W0 (+ A1 @ W1) + bias ) ----
#define BM 64
#define BN 32
#define BK 32
#define LDA 40   // bf16 elems, padded K stride for A tiles
#define LDT 40   // bf16 elems, padded K stride for transposed W tiles

__global__ __launch_bounds__(256)
void fused_linear_kernel(const float* __restrict__ A0, const float* __restrict__ A0den,
                         const float* __restrict__ A1,
                         const float* __restrict__ W0, const float* __restrict__ W1,
                         const float* __restrict__ bias,
                         void* __restrict__ outp,
                         int M, int K, int N, int out_bf16_relu)
{
  __shared__ bf16_t As0[BM * LDA];
  __shared__ bf16_t As1[BM * LDA];
  __shared__ bf16_t Wt0[BN * LDT];   // transposed: [n][k] so B frags are contiguous
  __shared__ bf16_t Wt1[BN * LDT];

  const int rowBase = blockIdx.y * BM;
  const int colBase = blockIdx.x * BN;
  const int wid  = threadIdx.x >> 5;
  const int lane = threadIdx.x & 31;
  const int wm = wid & 3;    // 4 row tiles of 16
  const int wn = wid >> 2;   // 2 col tiles of 16
  const bool dual = (A1 != nullptr);

  v8f acc = {};

  for (int k0 = 0; k0 < K; k0 += BK) {
    // ---- stage A tiles (64x32) via float4 loads, fp32 -> bf16, optional num/den fusion
    for (int i = threadIdx.x; i < (BM * BK) / 4; i += 256) {
      int m  = i >> 3;            // 8 float4 per row
      int kq = (i & 7) << 2;
      int row = rowBase + m;
      float4 v0 = make_float4(0.f, 0.f, 0.f, 0.f);
      float4 v1 = make_float4(0.f, 0.f, 0.f, 0.f);
      if (row < M) {
        long gi = (long)row * K + (k0 + kq);
        v0 = *(const float4*)&A0[gi];
        if (A0den) {
          float4 dd = *(const float4*)&A0den[gi];
          v0.x /= fmaxf(dd.x, 1e-16f); v0.y /= fmaxf(dd.y, 1e-16f);
          v0.z /= fmaxf(dd.z, 1e-16f); v0.w /= fmaxf(dd.w, 1e-16f);
        }
        if (dual) v1 = *(const float4*)&A1[gi];
      }
      bf16_t* a0p = &As0[m * LDA + kq];
      a0p[0] = (bf16_t)v0.x; a0p[1] = (bf16_t)v0.y; a0p[2] = (bf16_t)v0.z; a0p[3] = (bf16_t)v0.w;
      if (dual) {
        bf16_t* a1p = &As1[m * LDA + kq];
        a1p[0] = (bf16_t)v1.x; a1p[1] = (bf16_t)v1.y; a1p[2] = (bf16_t)v1.z; a1p[3] = (bf16_t)v1.w;
      }
    }
    // ---- stage W tiles (32x32) via float4 loads, store TRANSPOSED ([n][k])
    {
      int i  = threadIdx.x;       // exactly 256 float4 = 32x32 floats
      int kk = i >> 3;
      int nq = (i & 7) << 2;
      long gi = (long)(k0 + kk) * N + (colBase + nq);
      float4 w0 = *(const float4*)&W0[gi];
      Wt0[(nq + 0) * LDT + kk] = (bf16_t)w0.x;
      Wt0[(nq + 1) * LDT + kk] = (bf16_t)w0.y;
      Wt0[(nq + 2) * LDT + kk] = (bf16_t)w0.z;
      Wt0[(nq + 3) * LDT + kk] = (bf16_t)w0.w;
      if (dual) {
        float4 w1 = *(const float4*)&W1[gi];
        Wt1[(nq + 0) * LDT + kk] = (bf16_t)w1.x;
        Wt1[(nq + 1) * LDT + kk] = (bf16_t)w1.y;
        Wt1[(nq + 2) * LDT + kk] = (bf16_t)w1.z;
        Wt1[(nq + 3) * LDT + kk] = (bf16_t)w1.w;
      }
    }
    // prefetch next K-chunk of W while we compute this one (global_prefetch path)
    if (k0 + BK < K) {
      __builtin_prefetch(&W0[(long)(k0 + BK + (threadIdx.x >> 3)) * N + colBase + ((threadIdx.x & 7) << 2)], 0, 1);
    }
    __syncthreads();

    // ---- fragments (16-bit A 16x32 / B 32x16), k = 16*(lane/16) + i in both A and B (this code's order), all contiguous LDS reads
    const int kbA = (lane < 16) ? 0 : 16;   // A lane halves: K {0..15} / {16..31}
    const int kbB = (lane < 16) ? 0 : 16;   // B lane halves: K {0..15} / {16..31}
    const bf16_t* Ap0 = &As0[(wm * 16 + (lane & 15)) * LDA];
    const bf16_t* Bp0 = &Wt0[(wn * 16 + (lane & 15)) * LDT + kbB];

    v16bf a, b;
#pragma unroll
    for (int i = 0; i < 16; ++i) a[i] = Ap0[kbA + i];
#pragma unroll
    for (int i = 0; i < 16; ++i) b[i] = Bp0[i];
    acc = __builtin_amdgcn_wmma_f32_16x16x32_bf16(false, a, false, b, (short)0, acc, false, false);

    if (dual) {
      const bf16_t* Ap1 = &As1[(wm * 16 + (lane & 15)) * LDA];
      const bf16_t* Bp1 = &Wt1[(wn * 16 + (lane & 15)) * LDT + kbB];
      v16bf a1, b1;
#pragma unroll
      for (int i = 0; i < 16; ++i) a1[i] = Ap1[kbA + i];
#pragma unroll
      for (int i = 0; i < 16; ++i) b1[i] = Bp1[i];
      acc = __builtin_amdgcn_wmma_f32_16x16x32_bf16(false, a1, false, b1, (short)0, acc, false, false);
    }
    __syncthreads();
  }

  // epilogue: C layout = VGPR v -> M = v (lanes 0-15) / v+8 (lanes 16-31), N = lane&15
  const int col   = colBase + wn * 16 + (lane & 15);
  const int rbase = rowBase + wm * 16 + ((lane < 16) ? 0 : 8);
  const float bv  = bias ? bias[col] : 0.f;
  if (out_bf16_relu) {
    bf16_t* out = (bf16_t*)outp;
#pragma unroll
    for (int v = 0; v < 8; ++v) {
      int row = rbase + v;
      if (row < M) out[(long)row * N + col] = (bf16_t)fmaxf(acc[v] + bv, 0.f);
    }
  } else {
    float* out = (float*)outp;
#pragma unroll
    for (int v = 0; v < 8; ++v) {
      int row = rbase + v;
      if (row < M) out[(long)row * N + col] = acc[v] + bv;
    }
  }
}

// ---------------- edge pass 1: segment max of logits -----------------------------------
// one thread = one edge x 8 channels; messages gathered as a single 16B load
__global__ void edge_max_kernel(const int* __restrict__ ei, const bf16_t* __restrict__ xp,
                                const float* __restrict__ t, unsigned* __restrict__ menc,
                                int E, int Cbits)
{
  const int C = 1 << Cbits;
  const int cb = Cbits - 3;               // chunks of 8 channels
  long idx = (long)blockIdx.x * blockDim.x + threadIdx.x;
  if (idx >= ((long)E << cb)) return;
  int e  = (int)(idx >> cb);
  int c0 = (int)(idx & ((C >> 3) - 1)) << 3;
  int s = ei[e], d = ei[E + e];
  v8bf mv = *(const v8bf*)&xp[(long)s * C + c0];
  unsigned* mp = &menc[(long)d * C + c0];
#pragma unroll
  for (int i = 0; i < 8; ++i) {
    float logit = (float)mv[i] * t[c0 + i];
    atomicMax(&mp[i], enc_f32(logit));
  }
}

// ---------------- edge pass 2: denom += e, num += e*msg ---------------------------------
__global__ void edge_sum_kernel(const int* __restrict__ ei, const bf16_t* __restrict__ xp,
                                const float* __restrict__ t, const unsigned* __restrict__ menc,
                                float* __restrict__ den, float* __restrict__ num,
                                int E, int Cbits)
{
  const int C = 1 << Cbits;
  const int cb = Cbits - 3;
  long idx = (long)blockIdx.x * blockDim.x + threadIdx.x;
  if (idx >= ((long)E << cb)) return;
  int e  = (int)(idx >> cb);
  int c0 = (int)(idx & ((C >> 3) - 1)) << 3;
  int s = ei[e], d = ei[E + e];
  v8bf mv = *(const v8bf*)&xp[(long)s * C + c0];
  const uint4* mp = (const uint4*)&menc[(long)d * C + c0];
  uint4 m0 = mp[0], m1 = mp[1];
  unsigned mraw[8] = {m0.x, m0.y, m0.z, m0.w, m1.x, m1.y, m1.z, m1.w};
  float* dp = &den[(long)d * C + c0];
  float* np = &num[(long)d * C + c0];
#pragma unroll
  for (int i = 0; i < 8; ++i) {
    float msg = (float)mv[i];
    float ev  = __expf(msg * t[c0 + i] - dec_f32(mraw[i]));
    atomicAdd(&dp[i], ev);
    atomicAdd(&np[i], ev * msg);
  }
}

// ---------------- row L2-normalize + relu (in-place capable) ----------------------------
__global__ __launch_bounds__(128)
void rownorm_relu_kernel(const float* __restrict__ in, float* __restrict__ out, int rowlen)
{
  __shared__ float red[128];
  const long row = blockIdx.x;
  float ss = 0.f;
  for (int c = threadIdx.x; c < rowlen; c += 128) {
    float v = in[row * rowlen + c];
    ss += v * v;
  }
  red[threadIdx.x] = ss;
  __syncthreads();
  for (int s = 64; s > 0; s >>= 1) {
    if (threadIdx.x < s) red[threadIdx.x] += red[threadIdx.x + s];
    __syncthreads();
  }
  const float inv = 1.0f / fmaxf(sqrtf(red[0]), 1e-12f);
  for (int c = threadIdx.x; c < rowlen; c += 128) {
    out[row * rowlen + c] = fmaxf(in[row * rowlen + c] * inv, 0.f);
  }
}

// ---------------- MemPool: student-t assignment + S^T x (block-partial in LDS) ----------
__global__ __launch_bounds__(256)
void mempool_kernel(const float* __restrict__ h, const float* __restrict__ kmem,
                    const float* __restrict__ wconv, float* __restrict__ gpool, int n)
{
  __shared__ float keys[16 * 128];
  __shared__ float part[4 * 128];
  for (int i = threadIdx.x; i < 16 * 128; i += 256) keys[i] = kmem[i];
  for (int i = threadIdx.x; i < 4 * 128; i += 256) part[i] = 0.f;
  __syncthreads();

  const int node = blockIdx.x * 256 + threadIdx.x;
  float s[4] = {0.f, 0.f, 0.f, 0.f};
  const bool active = node < n;
  if (active) {
    float d2[16];
#pragma unroll
    for (int kk = 0; kk < 16; ++kk) d2[kk] = 0.f;
    for (int f = 0; f < 128; ++f) {
      float xv = h[(long)node * 128 + f];
#pragma unroll
      for (int kk = 0; kk < 16; ++kk) {
        float df = keys[kk * 128 + f] - xv;
        d2[kk] += df * df;
      }
    }
    float mixed[4] = {0.f, 0.f, 0.f, 0.f};
#pragma unroll
    for (int hh = 0; hh < 4; ++hh) {
      float dist[4], hs = 0.f;
#pragma unroll
      for (int k = 0; k < 4; ++k) { dist[k] = 1.0f / (1.0f + fmaxf(d2[hh * 4 + k], 0.f)); hs += dist[k]; }
      float wc = wconv[hh] / hs;
#pragma unroll
      for (int k = 0; k < 4; ++k) mixed[k] += wc * dist[k];
    }
    float mx = fmaxf(fmaxf(mixed[0], mixed[1]), fmaxf(mixed[2], mixed[3]));
    float se = 0.f;
#pragma unroll
    for (int k = 0; k < 4; ++k) { s[k] = __expf(mixed[k] - mx); se += s[k]; }
    float inv = 1.0f / se;
#pragma unroll
    for (int k = 0; k < 4; ++k) s[k] *= inv;

    for (int f = 0; f < 128; ++f) {
      float xv = h[(long)node * 128 + f];
#pragma unroll
      for (int k = 0; k < 4; ++k) atomicAdd(&part[k * 128 + f], s[k] * xv);  // ds_add_f32
    }
  }
  __syncthreads();
  for (int i = threadIdx.x; i < 4 * 128; i += 256) atomicAdd(&gpool[i], part[i]);
}

// ---------------- head: memlin + mean-pool + fx + layernorm + relu ----------------------
__global__ __launch_bounds__(128)
void head_kernel(const float* __restrict__ gpool, const float* __restrict__ wml,
                 const float* __restrict__ bml, const float* __restrict__ wfx,
                 const float* __restrict__ bfx, const float* __restrict__ gamma,
                 const float* __restrict__ beta, float* __restrict__ out)
{
  __shared__ float g[128];
  __shared__ float y[64];
  __shared__ float mu_s, rstd_s;
  const int f = threadIdx.x;

  {
    float acc0 = 0.f, acc1 = 0.f, acc2 = 0.f, acc3 = 0.f;
    for (int j = 0; j < 128; ++j) {
      float w = wml[j * 128 + f];
      acc0 += gpool[0 * 128 + j] * w;
      acc1 += gpool[1 * 128 + j] * w;
      acc2 += gpool[2 * 128 + j] * w;
      acc3 += gpool[3 * 128 + j] * w;
    }
    g[f] = 0.25f * (acc0 + acc1 + acc2 + acc3) + bml[f];
  }
  __syncthreads();
  if (f < 64) {
    float acc = bfx[f];
    for (int j = 0; j < 128; ++j) acc += g[j] * wfx[j * 64 + f];
    y[f] = acc;
  }
  __syncthreads();
  if (f == 0) {
    float mu = 0.f;
    for (int i = 0; i < 64; ++i) mu += y[i];
    mu *= (1.0f / 64.0f);
    float var = 0.f;
    for (int i = 0; i < 64; ++i) { float d = y[i] - mu; var += d * d; }
    var *= (1.0f / 64.0f);
    mu_s = mu;
    rstd_s = rsqrtf(var + 1e-5f);
  }
  __syncthreads();
  if (f < 64) out[f] = fmaxf((y[f] - mu_s) * rstd_s * gamma[f] + beta[f], 0.f);
}

// ---------------- driver ----------------------------------------------------------------
extern "C" void kernel_launch(void* const* d_in, const int* in_sizes, int n_in,
                              void* d_out, int out_size, void* d_ws, size_t ws_size,
                              hipStream_t stream)
{
  (void)in_sizes; (void)n_in; (void)out_size; (void)ws_size;
  const float* x        = (const float*)d_in[0];
  const int*   ei       = (const int*)  d_in[1];
  const float* w_proj1  = (const float*)d_in[2];
  const float* b_proj1  = (const float*)d_in[3];
  const float* t1       = (const float*)d_in[4];
  const float* w_l1     = (const float*)d_in[5];
  const float* b_l1     = (const float*)d_in[6];
  const float* w_r1     = (const float*)d_in[7];
  const float* w_proj2  = (const float*)d_in[8];
  const float* b_proj2  = (const float*)d_in[9];
  const float* t2       = (const float*)d_in[10];
  const float* w_l2     = (const float*)d_in[11];
  const float* b_l2     = (const float*)d_in[12];
  const float* w_r2     = (const float*)d_in[13];
  const float* k_mem    = (const float*)d_in[14];
  const float* w_conv   = (const float*)d_in[15];
  const float* w_memlin = (const float*)d_in[16];
  const float* b_memlin = (const float*)d_in[17];
  const float* w_fx     = (const float*)d_in[18];
  const float* b_fx     = (const float*)d_in[19];
  const float* gamma    = (const float*)d_in[20];
  const float* beta     = (const float*)d_in[21];
  float* out = (float*)d_out;

  // ---- workspace carve (bytes) ----
  char* ws = (char*)d_ws;
  size_t off = 0;
  const size_t NODE_C = (size_t)N_NODES * 256;
  auto carve = [&](size_t bytes) { size_t r = off; off += (bytes + 255) & ~(size_t)255; return r; };
  bf16_t*   xp   = (bf16_t*)  (ws + carve(NODE_C * sizeof(bf16_t)));   // projected messages (bf16)
  unsigned* menc = (unsigned*)(ws + carve(NODE_C * sizeof(unsigned))); // segment max (ordered uint)
  float*    den  = (float*)   (ws + carve(NODE_C * sizeof(float)));    // softmax denom
  float*    num  = (float*)   (ws + carve(NODE_C * sizeof(float)));    // softmax numer
  float*    bufA = (float*)   (ws + carve(NODE_C * sizeof(float)));    // h1 (50000x256)
  float*    bufB = (float*)   (ws + carve((size_t)N_NODES * 128 * sizeof(float))); // h2
  float*    gpool= (float*)   (ws + carve(4 * 128 * sizeof(float)));   // S^T x accumulator

  const int gy = (N_NODES + BM - 1) / BM;  // 782

  // ================= Layer 1 (C = 64) =================
  {
    const long n4 = ((long)N_NODES * 64) / 4;
    fill_u32_kernel<<<(int)((n4 + 255) / 256), 256, 0, stream>>>((uint4*)menc, ENC_NEG_INF, n4);
    fill_u32_kernel<<<(int)((n4 + 255) / 256), 256, 0, stream>>>((uint4*)den, 0u, n4);
    fill_u32_kernel<<<(int)((n4 + 255) / 256), 256, 0, stream>>>((uint4*)num, 0u, n4);

    fused_linear_kernel<<<dim3(64 / BN, gy), 256, 0, stream>>>(
        x, nullptr, nullptr, w_proj1, nullptr, b_proj1, xp, N_NODES, 64, 64, 1);

    const long tot = (long)N_EDGES << 3;   // E * (64/8)
    edge_max_kernel<<<(int)((tot + 255) / 256), 256, 0, stream>>>(ei, xp, t1, menc, N_EDGES, 6);
    edge_sum_kernel<<<(int)((tot + 255) / 256), 256, 0, stream>>>(ei, xp, t1, menc, den, num, N_EDGES, 6);

    fused_linear_kernel<<<dim3(256 / BN, gy), 256, 0, stream>>>(
        num, den, x, w_l1, w_r1, b_l1, bufA, N_NODES, 64, 256, 0);
    rownorm_relu_kernel<<<N_NODES, 128, 0, stream>>>(bufA, bufA, 256);
  }

  // ================= Layer 2 (C = 256) =================
  {
    const long n4 = ((long)N_NODES * 256) / 4;
    fill_u32_kernel<<<(int)((n4 + 255) / 256), 256, 0, stream>>>((uint4*)menc, ENC_NEG_INF, n4);
    fill_u32_kernel<<<(int)((n4 + 255) / 256), 256, 0, stream>>>((uint4*)den, 0u, n4);
    fill_u32_kernel<<<(int)((n4 + 255) / 256), 256, 0, stream>>>((uint4*)num, 0u, n4);

    fused_linear_kernel<<<dim3(256 / BN, gy), 256, 0, stream>>>(
        bufA, nullptr, nullptr, w_proj2, nullptr, b_proj2, xp, N_NODES, 256, 256, 1);

    const long tot = (long)N_EDGES << 5;   // E * (256/8)
    edge_max_kernel<<<(int)((tot + 255) / 256), 256, 0, stream>>>(ei, xp, t2, menc, N_EDGES, 8);
    edge_sum_kernel<<<(int)((tot + 255) / 256), 256, 0, stream>>>(ei, xp, t2, menc, den, num, N_EDGES, 8);

    fused_linear_kernel<<<dim3(128 / BN, gy), 256, 0, stream>>>(
        num, den, bufA, w_l2, w_r2, b_l2, bufB, N_NODES, 256, 128, 0);
    rownorm_relu_kernel<<<N_NODES, 128, 0, stream>>>(bufB, bufB, 128);
  }

  // ================= MemPool + head =================
  fill_u32_kernel<<<1, 128, 0, stream>>>((uint4*)gpool, 0u, (4 * 128) / 4);
  mempool_kernel<<<(N_NODES + 255) / 256, 256, 0, stream>>>(bufB, k_mem, w_conv, gpool, N_NODES);
  head_kernel<<<1, 128, 0, stream>>>(gpool, w_memlin, b_memlin, w_fx, b_fx, gamma, beta, out);
}
